// GatedGINConv_37391985278998
// MI455X (gfx1250) — compile-verified, matrix operand order checked by reading, not run on hardware
//
#include <hip/hip_runtime.h>
#include <hip/hip_bf16.h>

// ---------------------------------------------------------------------------
// CDNA5 (gfx1250) GatedGIN forward. GEMMs via v_wmma_f32_16x16x32_bf16,
// bf16 operands / fp32 accumulate, double-buffered LDS staging through
// GLOBAL_LOAD_ASYNC_TO_LDS_B128 (ASYNCcnt) with s_wait_asynccnt.
// ---------------------------------------------------------------------------

typedef __bf16 bf16_t;
typedef __attribute__((ext_vector_type(16))) __bf16 v16bf;
typedef __attribute__((ext_vector_type(8)))  float  v8f;

__device__ inline bf16_t f2bf(float x) { return (bf16_t)x; }

// ------------------------------ WMMA GEMM ----------------------------------
// C[M x Nc] = A[M x K](bf16, row-major) * Bt[Nc x K](bf16 = W^T) + bias
#define BM 128
#define BN 128
#define BK 32
#define LDT (BK + 8)   // padded LDS row (bf16 elements)

__device__ inline void async_load16(unsigned lds_addr, const void* gaddr) {
    // GLOBAL_LOAD_ASYNC_TO_LDS_B128: VDST = LDS byte address, VADDR = 64-bit VA
    asm volatile("global_load_async_to_lds_b128 %0, %1, off"
                 :: "v"(lds_addr), "v"(gaddr) : "memory");
}
__device__ inline void wait_async0() {
    asm volatile("s_wait_asynccnt 0x0" ::: "memory");
}

__device__ inline v16bf lds_frag(const bf16_t* __restrict__ p, int half) {
    v16bf f;
#pragma unroll
    for (int j = 0; j < 8; ++j) {
        int k = ((j >= 4) ? 16 : 0) + half * 8 + (j & 3) * 2;
        f[2 * j]     = p[k];
        f[2 * j + 1] = p[k + 1];
    }
    return f;
}

__global__ __launch_bounds__(256) void wmma_gemm_bias_kernel(
    const bf16_t* __restrict__ A, const bf16_t* __restrict__ Bt,
    const float* __restrict__ bias, float* __restrict__ C,
    int Mrows, int Ncols, int K)
{
    __shared__ bf16_t Asb[2][BM * LDT];
    __shared__ bf16_t Bsb[2][BN * LDT];

    const int tid   = threadIdx.x;
    const int lane  = tid & 31;
    const int wave  = tid >> 5;      // 0..7
    const int waveM = wave >> 1;     // 0..3 : 32-row slab
    const int waveN = wave & 1;      // 0..1 : 64-col slab
    const int half  = lane >> 4;
    const int r     = lane & 15;

    const int blockRow = blockIdx.x * BM;
    const int blockCol = blockIdx.y * BN;
    const bool fullM = (blockRow + BM <= Mrows);
    const bool fullN = (blockCol + BN <= Ncols);

    // tile-load mapping: 256 threads, 16 bf16 (two b128 chunks) each
    const int lrow = tid >> 1;
    const int lkk  = (tid & 1) * 16;

    auto load_tiles = [&](int k0, int buf) {
        // ---- A tile: BM x BK ----
        {
            const int grow = blockRow + lrow;
            bf16_t* dst = &Asb[buf][lrow * LDT + lkk];
            if (fullM) {
                const bf16_t* src = A + (size_t)grow * K + k0 + lkk;
                async_load16((unsigned)(uintptr_t)(void*)dst, src);
                async_load16((unsigned)(uintptr_t)(void*)(dst + 8), src + 8);
            } else {
                uint4 v0 = make_uint4(0u, 0u, 0u, 0u), v1 = v0;
                if (grow < Mrows) {
                    const uint4* s = (const uint4*)(A + (size_t)grow * K + k0 + lkk);
                    v0 = s[0]; v1 = s[1];
                }
                *(uint4*)dst       = v0;
                *(uint4*)(dst + 8) = v1;
            }
        }
        // ---- B tile (W^T): BN x BK ----
        {
            const int gcol = blockCol + lrow;
            bf16_t* dst = &Bsb[buf][lrow * LDT + lkk];
            if (fullN) {
                const bf16_t* src = Bt + (size_t)gcol * K + k0 + lkk;
                async_load16((unsigned)(uintptr_t)(void*)dst, src);
                async_load16((unsigned)(uintptr_t)(void*)(dst + 8), src + 8);
            } else {
                uint4 v0 = make_uint4(0u, 0u, 0u, 0u), v1 = v0;
                if (gcol < Ncols) {
                    const uint4* s = (const uint4*)(Bt + (size_t)gcol * K + k0 + lkk);
                    v0 = s[0]; v1 = s[1];
                }
                *(uint4*)dst       = v0;
                *(uint4*)(dst + 8) = v1;
            }
        }
    };

    v8f acc[2][4];
#pragma unroll
    for (int i = 0; i < 2; ++i)
#pragma unroll
        for (int j = 0; j < 4; ++j)
            acc[i][j] = (v8f){0.f, 0.f, 0.f, 0.f, 0.f, 0.f, 0.f, 0.f};

    // prologue
    load_tiles(0, 0);
    wait_async0();
    __syncthreads();

    int cur = 0;
    for (int k0 = 0; k0 < K; k0 += BK) {
        if (k0 + BK < K)
            load_tiles(k0 + BK, cur ^ 1);   // stream next slab into other buffer

        const bf16_t* Ab = Asb[cur];
        const bf16_t* Bb = Bsb[cur];
        v16bf fa[2], fb[4];
#pragma unroll
        for (int i = 0; i < 2; ++i)
            fa[i] = lds_frag(Ab + (waveM * 32 + i * 16 + r) * LDT, half);
#pragma unroll
        for (int j = 0; j < 4; ++j)
            fb[j] = lds_frag(Bb + (waveN * 64 + j * 16 + r) * LDT, half);

#pragma unroll
        for (int i = 0; i < 2; ++i)
#pragma unroll
            for (int j = 0; j < 4; ++j)
                acc[i][j] = __builtin_amdgcn_wmma_f32_16x16x32_bf16(
                    false, fa[i], false, fb[j], (short)0, acc[i][j], false, false);

        wait_async0();      // next-slab async writes to LDS complete
        __syncthreads();    // all waves done reading cur + done writing nxt
        cur ^= 1;
    }

#pragma unroll
    for (int i = 0; i < 2; ++i)
#pragma unroll
        for (int j = 0; j < 4; ++j) {
            const int tr  = blockRow + waveM * 32 + i * 16;
            const int col = blockCol + waveN * 64 + j * 16 + r;
            if (col < Ncols) {
                const float bv = bias ? bias[col] : 0.f;
#pragma unroll
                for (int v = 0; v < 8; ++v) {
                    const int row = tr + v + half * 8;
                    if (row < Mrows)
                        C[(size_t)row * Ncols + col] = acc[i][j][v] + bv;
                }
            }
        }
}

// --------------------------- elementwise kernels ---------------------------

__global__ void cast_bf16_kernel(const float* __restrict__ X, bf16_t* __restrict__ Y, long n) {
    long i = (long)blockIdx.x * blockDim.x + threadIdx.x;
    if (i < n) Y[i] = f2bf(X[i]);
}

__global__ void transpose_cast_kernel(const float* __restrict__ W, bf16_t* __restrict__ Wt, int Dd) {
    long i = (long)blockIdx.x * blockDim.x + threadIdx.x;
    if (i >= (long)Dd * Dd) return;
    int nn = (int)(i / Dd), kk = (int)(i % Dd);
    Wt[i] = f2bf(W[(size_t)kk * Dd + nn]);
}

__global__ void zero_kernel(float* __restrict__ X, long n) {
    long i = (long)blockIdx.x * blockDim.x + threadIdx.x;
    if (i < n) X[i] = 0.f;
}

__global__ void e_pre_kernel(const float* __restrict__ Be, const float* __restrict__ Ah,
                             const float* __restrict__ Cu, const int* __restrict__ bond_atoms,
                             const int* __restrict__ bond_graph, const float* __restrict__ eps,
                             float* __restrict__ out, int M, int D) {
    long i = (long)blockIdx.x * blockDim.x + threadIdx.x;
    if (i >= (long)M * D) return;
    int m = (int)(i / D), d = (int)(i % D);
    float e1p = 1.f + eps[0];
    float be = Be[i];
    float s  = e1p * be;
    int a0 = bond_atoms[2 * m], a1 = bond_atoms[2 * m + 1];
    float v = fmaxf(s + Ah[(size_t)a0 * D + d] + Ah[(size_t)a1 * D + d], 0.f)
            + fmaxf(s + be, 0.f)
            + fmaxf(s + Cu[(size_t)bond_graph[m] * D + d], 0.f);
    out[i] = v;
}

__global__ void col_stats_kernel(const float* __restrict__ X, int rows, int D,
                                 float* __restrict__ sum, float* __restrict__ sumsq,
                                 int rowsPerBlock) {
    int d = blockIdx.x * blockDim.x + threadIdx.x;
    if (d >= D) return;
    int r0 = blockIdx.y * rowsPerBlock;
    int r1 = min(r0 + rowsPerBlock, rows);
    float s = 0.f, q = 0.f;
    for (int r = r0; r < r1; ++r) {
        float v = X[(size_t)r * D + d];
        s += v; q += v * v;
    }
    atomicAdd(&sum[d], s);
    atomicAdd(&sumsq[d], q);
}

__global__ void bn_relu_kernel(float* __restrict__ X, int rows, int D,
                               const float* __restrict__ sum, const float* __restrict__ sumsq,
                               const float* __restrict__ g, const float* __restrict__ b) {
    long i = (long)blockIdx.x * blockDim.x + threadIdx.x;
    if (i >= (long)rows * D) return;
    int d = (int)(i % D);
    float m  = sum[d] / (float)rows;
    float vv = sumsq[d] / (float)rows - m * m;
    float y  = g[d] * (X[i] - m) * rsqrtf(vv + 1e-5f) + b[d];
    X[i] = fmaxf(y, 0.f);
}

__global__ void gated_scatter_kernel(const float* __restrict__ e_new, const float* __restrict__ Eh,
                                     const int* __restrict__ bond_atoms,
                                     float* __restrict__ num, float* __restrict__ den,
                                     int M, int D) {
    long i = (long)blockIdx.x * blockDim.x + threadIdx.x;
    if (i >= (long)M * D) return;
    int m = (int)(i / D), d = (int)(i % D);
    float sg = 1.f / (1.f + __expf(-e_new[i]));
    int a0 = bond_atoms[2 * m], a1 = bond_atoms[2 * m + 1];
    atomicAdd(&num[(size_t)a0 * D + d], sg * Eh[(size_t)a1 * D + d]);
    atomicAdd(&den[(size_t)a0 * D + d], sg);
    atomicAdd(&num[(size_t)a1 * D + d], sg * Eh[(size_t)a0 * D + d]);
    atomicAdd(&den[(size_t)a1 * D + d], sg);
}

__global__ void h_new_kernel(const float* __restrict__ Ah, const float* __restrict__ Dh,
                             const float* __restrict__ Fu, const int* __restrict__ atom_graph,
                             const float* __restrict__ num, const float* __restrict__ den,
                             const float* __restrict__ eps, float* __restrict__ out,
                             int N, int D) {
    long i = (long)blockIdx.x * blockDim.x + threadIdx.x;
    if (i >= (long)N * D) return;
    int n = (int)(i / D), d = (int)(i % D);
    float e1p = 1.f + eps[0];
    float ah  = Ah[i];
    float v = fmaxf(e1p * ah + Dh[i], 0.f)
            + fmaxf(e1p * ah + Fu[(size_t)atom_graph[n] * D + d], 0.f)
            + num[i] / (den[i] + 1e-6f);
    out[i] = v;
}

__global__ void c12_cast_kernel(const float* __restrict__ h1n, const float* __restrict__ h2n,
                                bf16_t* __restrict__ x1b, bf16_t* __restrict__ x2b, long n) {
    long i = (long)blockIdx.x * blockDim.x + threadIdx.x;
    if (i >= n) return;
    float a = h1n[i], b = h2n[i];
    float c = 1.f / (fabsf(a - b) + 1.f);
    x1b[i] = f2bf(c * a);
    x2b[i] = f2bf(c * b);
}

__global__ void mix_kernel(const float* __restrict__ hn, const float* __restrict__ m,
                           float* __restrict__ out, long n) {
    long i = (long)blockIdx.x * blockDim.x + threadIdx.x;
    if (i < n) out[i] = 0.8f * hn[i] + 0.2f * m[i];
}

__global__ void count_kernel(const int* __restrict__ seg, float* __restrict__ cnt, int n) {
    int i = blockIdx.x * blockDim.x + threadIdx.x;
    if (i < n) atomicAdd(&cnt[seg[i]], 1.f);
}

__global__ void segsum_kernel(const float* __restrict__ X, const int* __restrict__ seg,
                              float* __restrict__ acc, int rows, int D) {
    long i = (long)blockIdx.x * blockDim.x + threadIdx.x;
    if (i >= (long)rows * D) return;
    int r = (int)(i / D), d = (int)(i % D);
    atomicAdd(&acc[(size_t)seg[r] * D + d], X[i]);
}

__global__ void u_new_kernel(const float* __restrict__ Iu,
                             const float* __restrict__ a1, const float* __restrict__ a2,
                             const float* __restrict__ b1, const float* __restrict__ b2,
                             const float* __restrict__ cnta, const float* __restrict__ cntb,
                             const float* __restrict__ eps, float* __restrict__ out,
                             int Bg, int D) {
    long i = (long)blockIdx.x * blockDim.x + threadIdx.x;
    if (i >= (long)Bg * D) return;
    int g = (int)(i / D);
    float e1p = 1.f + eps[0];
    float iu = Iu[i];
    float ca = fmaxf(cnta[g], 1.f), cb = fmaxf(cntb[g], 1.f);
    out[i] = fmaxf(e1p * iu + a1[i] / ca, 0.f)
           + fmaxf(e1p * iu + b1[i] / cb, 0.f)
           + fmaxf(e1p * iu + a2[i] / ca, 0.f)
           + fmaxf(e1p * iu + b2[i] / cb, 0.f);
}

// ------------------------------- launcher ----------------------------------

static inline int cdiv(long a, long b) { return (int)((a + b - 1) / b); }

extern "C" void kernel_launch(void* const* d_in, const int* in_sizes, int n_in,
                              void* d_out, int out_size, void* d_ws, size_t ws_size,
                              hipStream_t stream) {
    (void)n_in; (void)out_size; (void)ws_size;
    const int D = 800;
    const int N  = in_sizes[0] / D;
    const int M  = in_sizes[1] / D;
    const int Bg = in_sizes[4] / D;

    const float* h   = (const float*)d_in[0];
    const float* e   = (const float*)d_in[1];
    const float* h2  = (const float*)d_in[2];
    const float* e2  = (const float*)d_in[3];
    const float* u   = (const float*)d_in[4];
    const int* bond_atoms = (const int*)d_in[5];
    const int* bond_graph = (const int*)d_in[6];
    const int* atom_graph = (const int*)d_in[7];
    // params: WA,bA, WB,bB, WC,bC, WD,bD, WE,bE, WF,bF, WG,bG, WH,bH, WI,bI, Ww,bw
    const float* W[10]; const float* bias[10];
    for (int i = 0; i < 10; ++i) { W[i] = (const float*)d_in[8 + 2 * i]; bias[i] = (const float*)d_in[9 + 2 * i]; }
    const float* eps    = (const float*)d_in[28];
    const float* bn_g_h = (const float*)d_in[29];
    const float* bn_b_h = (const float*)d_in[30];
    const float* bn_g_e = (const float*)d_in[31];
    const float* bn_b_e = (const float*)d_in[32];
    const float* bn_g_u = (const float*)d_in[33];
    const float* bn_b_u = (const float*)d_in[34];

    // output slots
    float* out   = (float*)d_out;
    float* o_h1f = out;
    float* o_e1  = o_h1f + (size_t)N * D;
    float* o_h2f = o_e1  + (size_t)M * D;
    float* o_e2  = o_h2f + (size_t)N * D;
    float* o_u   = o_e2  + (size_t)M * D;

    // workspace bump allocator
    char* wp = (char*)d_ws;
    auto alloc = [&](size_t bytes) -> void* {
        void* p = (void*)wp;
        wp += (bytes + 255) & ~(size_t)255;
        return p;
    };
    bf16_t* Wt[10];
    for (int i = 0; i < 10; ++i) Wt[i] = (bf16_t*)alloc((size_t)D * D * sizeof(bf16_t));
    bf16_t* hb  = (bf16_t*)alloc((size_t)N * D * sizeof(bf16_t));
    bf16_t* h2b = (bf16_t*)alloc((size_t)N * D * sizeof(bf16_t));
    bf16_t* eb  = (bf16_t*)alloc((size_t)M * D * sizeof(bf16_t));
    bf16_t* e2b = (bf16_t*)alloc((size_t)M * D * sizeof(bf16_t));
    bf16_t* ub  = (bf16_t*)alloc((size_t)Bg * D * sizeof(bf16_t));
    float* Cu = (float*)alloc((size_t)Bg * D * 4);
    float* Fu = (float*)alloc((size_t)Bg * D * 4);
    float* Iu = (float*)alloc((size_t)Bg * D * 4);
    float* Ah = (float*)alloc((size_t)N * D * 4);
    float* Dh = (float*)alloc((size_t)N * D * 4);
    float* Eh = (float*)alloc((size_t)N * D * 4);
    float* Be = (float*)alloc((size_t)M * D * 4);
    float* num = (float*)alloc((size_t)N * D * 4);
    float* den = (float*)alloc((size_t)N * D * 4);
    float* h1n = (float*)alloc((size_t)N * D * 4);
    float* h2n = (float*)alloc((size_t)N * D * 4);
    float* s_sum = (float*)alloc((size_t)D * 4);
    float* s_sq  = (float*)alloc((size_t)D * 4);
    float* acc0 = (float*)alloc((size_t)Bg * D * 4);
    float* acc1 = (float*)alloc((size_t)Bg * D * 4);
    float* acc2 = (float*)alloc((size_t)Bg * D * 4);
    float* acc3 = (float*)alloc((size_t)Bg * D * 4);
    float* cnta = (float*)alloc((size_t)Bg * 4);
    float* cntb = (float*)alloc((size_t)Bg * 4);

    const int TB = 256;
    const long nND = (long)N * D, nMD = (long)M * D, nBD = (long)Bg * D;

    auto gemm = [&](const bf16_t* A, int widx, float* Cmat, int rows) {
        dim3 grid(cdiv(rows, BM), cdiv(D, BN));
        wmma_gemm_bias_kernel<<<grid, 256, 0, stream>>>(A, Wt[widx], bias[widx], Cmat, rows, D, D);
    };
    auto zero = [&](float* p, long n) { zero_kernel<<<cdiv(n, TB), TB, 0, stream>>>(p, n); };
    auto bn = [&](float* X, int rows, const float* g, const float* b) {
        zero(s_sum, D); zero(s_sq, D);
        dim3 sg(cdiv(D, TB), cdiv(rows, 256));
        col_stats_kernel<<<sg, TB, 0, stream>>>(X, rows, D, s_sum, s_sq, 256);
        bn_relu_kernel<<<cdiv((long)rows * D, TB), TB, 0, stream>>>(X, rows, D, s_sum, s_sq, g, b);
    };

    // ---- stage 0: casts ----
    for (int i = 0; i < 10; ++i)
        transpose_cast_kernel<<<cdiv((long)D * D, TB), TB, 0, stream>>>(W[i], Wt[i], D);
    cast_bf16_kernel<<<cdiv(nND, TB), TB, 0, stream>>>(h,  hb,  nND);
    cast_bf16_kernel<<<cdiv(nND, TB), TB, 0, stream>>>(h2, h2b, nND);
    cast_bf16_kernel<<<cdiv(nMD, TB), TB, 0, stream>>>(e,  eb,  nMD);
    cast_bf16_kernel<<<cdiv(nMD, TB), TB, 0, stream>>>(e2, e2b, nMD);
    cast_bf16_kernel<<<cdiv(nBD, TB), TB, 0, stream>>>(u,  ub,  nBD);

    // ---- stage 1: global-feature GEMMs ----
    gemm(ub, 2, Cu, Bg);  // WC
    gemm(ub, 5, Fu, Bg);  // WF
    gemm(ub, 8, Iu, Bg);  // WI

    // ---- stage 2: both views ----
    for (int v = 0; v < 2; ++v) {
        const bf16_t* hvb = v ? h2b : hb;
        const bf16_t* evb = v ? e2b : eb;
        float* e_out = v ? o_e2 : o_e1;
        float* h_out = v ? h2n : h1n;

        gemm(hvb, 0, Ah, N);  // WA
        gemm(hvb, 3, Dh, N);  // WD
        gemm(hvb, 4, Eh, N);  // WE
        gemm(evb, 1, Be, M);  // WB

        e_pre_kernel<<<cdiv(nMD, TB), TB, 0, stream>>>(Be, Ah, Cu, bond_atoms, bond_graph, eps, e_out, M, D);
        bn(e_out, M, bn_g_e, bn_b_e);   // e_out is now e_new (output e1 / e2n)

        zero(num, nND); zero(den, nND);
        gated_scatter_kernel<<<cdiv(nMD, TB), TB, 0, stream>>>(e_out, Eh, bond_atoms, num, den, M, D);
        h_new_kernel<<<cdiv(nND, TB), TB, 0, stream>>>(Ah, Dh, Fu, atom_graph, num, den, eps, h_out, N, D);
    }

    // ---- stage 3: cross-view mixing ----
    c12_cast_kernel<<<cdiv(nND, TB), TB, 0, stream>>>(h1n, h2n, hb, h2b, nND); // hb=x1b, h2b=x2b
    gemm(h2b, 9, Ah, N);   // m21 = (c12*h2n) @ Ww
    mix_kernel<<<cdiv(nND, TB), TB, 0, stream>>>(h1n, Ah, o_h1f, nND);
    gemm(hb, 9, Ah, N);    // m12 = (c12*h1n) @ Ww
    mix_kernel<<<cdiv(nND, TB), TB, 0, stream>>>(h2n, Ah, o_h2f, nND);
    bn(o_h1f, N, bn_g_h, bn_b_h);
    bn(o_h2f, N, bn_g_h, bn_b_h);

    // ---- stage 4: readout GEMMs + segment means ----
    cast_bf16_kernel<<<cdiv(nND, TB), TB, 0, stream>>>(o_h1f, hb,  nND);
    cast_bf16_kernel<<<cdiv(nND, TB), TB, 0, stream>>>(o_h2f, h2b, nND);
    cast_bf16_kernel<<<cdiv(nMD, TB), TB, 0, stream>>>(o_e1,  eb,  nMD);
    cast_bf16_kernel<<<cdiv(nMD, TB), TB, 0, stream>>>(o_e2,  e2b, nMD);

    zero(cnta, Bg); zero(cntb, Bg);
    count_kernel<<<cdiv(N, TB), TB, 0, stream>>>(atom_graph, cnta, N);
    count_kernel<<<cdiv(M, TB), TB, 0, stream>>>(bond_graph, cntb, M);
    zero(acc0, nBD); zero(acc1, nBD); zero(acc2, nBD); zero(acc3, nBD);

    gemm(hb, 6, Ah, N);   // Gh1
    segsum_kernel<<<cdiv(nND, TB), TB, 0, stream>>>(Ah, atom_graph, acc0, N, D);
    gemm(h2b, 6, Ah, N);  // Gh2
    segsum_kernel<<<cdiv(nND, TB), TB, 0, stream>>>(Ah, atom_graph, acc1, N, D);
    gemm(eb, 7, Be, M);   // He1
    segsum_kernel<<<cdiv(nMD, TB), TB, 0, stream>>>(Be, bond_graph, acc2, M, D);
    gemm(e2b, 7, Be, M);  // He2
    segsum_kernel<<<cdiv(nMD, TB), TB, 0, stream>>>(Be, bond_graph, acc3, M, D);

    u_new_kernel<<<cdiv(nBD, TB), TB, 0, stream>>>(Iu, acc0, acc1, acc2, acc3, cnta, cntb, eps, o_u, Bg, D);
    bn(o_u, Bg, bn_g_u, bn_b_u);
}
